// EdgePooling_67740224193148
// MI455X (gfx1250) — matrix operand order repaired and checked by reading, not compiled, not run on hardware
//
#include <hip/hip_runtime.h>
#include <hip/hip_bf16.h>
#include <stdint.h>

typedef __attribute__((ext_vector_type(16))) _Float16 v16h;
typedef __attribute__((ext_vector_type(8)))  float    v8f;

// ---------- small helpers ----------
__device__ inline unsigned f2o(float f) {            // monotonic float->uint
    unsigned u = __float_as_uint(f);
    return (u & 0x80000000u) ? ~u : (u | 0x80000000u);
}
__device__ inline float o2f(unsigned o) {
    unsigned u = (o & 0x80000000u) ? (o & 0x7FFFFFFFu) : ~o;
    return __uint_as_float(u);
}

// ---------- init / zero ----------
__global__ void zero_f32(float* p, long long n) {
    long long i = (long long)blockIdx.x * blockDim.x + threadIdx.x;
    long long stride = (long long)gridDim.x * blockDim.x;
    for (; i < n; i += stride) p[i] = 0.0f;
}
__global__ void init_node_state(unsigned* mmax, float* ssum, int* rem, int* clus,
                                float* nes, int N) {
    int i = blockIdx.x * blockDim.x + threadIdx.x;
    if (i < N) { mmax[i] = 0u; ssum[i] = 0.0f; rem[i] = 1; clus[i] = 0; nes[i] = 1.0f; }
}

// ---------- 1. edge scores: raw = concat(x[s],x[d],ea[e]) . w + b  (WMMA) ----------
// Block = 1 wave (32 lanes) handles a tile of 16 edges. C==128 path.
// Feature rows are staged into LDS with CDNA5 async-to-LDS DMA (ASYNCcnt):
// one global_load_async_to_lds_b128 moves 16B/lane = 512B/row-segment with no
// VGPR round trip. A tile: 16x384 f16 (12 chunks of 16x32). B: 32x16 with the
// w-chunk replicated in every column so every D column equals the dot product.
__global__ __launch_bounds__(32) void score_wmma(const float* __restrict__ x,
                                                 const int* __restrict__ ei,
                                                 const float* __restrict__ ea,
                                                 const float* __restrict__ lw,
                                                 const float* __restrict__ lb,
                                                 float* __restrict__ raw, int E) {
    __shared__ __align__(16) float feat[16][388];   // 16 edges x 384 f32 (+pad)
    const int base = blockIdx.x * 16;
    const int l = threadIdx.x;

    for (int m = 0; m < 16; ++m) {
        int e = base + m;
        if (e < E) {                                 // block-uniform branch
            int s = ei[e], d = ei[E + e];
            unsigned long long gx = (unsigned long long)(uintptr_t)(x  + (size_t)s * 128 + l * 4);
            unsigned long long gd = (unsigned long long)(uintptr_t)(x  + (size_t)d * 128 + l * 4);
            unsigned long long ge = (unsigned long long)(uintptr_t)(ea + (size_t)e * 128 + l * 4);
            unsigned l0 = (unsigned)(uintptr_t)&feat[m][l * 4];         // LDS byte offset
            unsigned l1 = (unsigned)(uintptr_t)&feat[m][128 + l * 4];   // (flat addr[31:0])
            unsigned l2 = (unsigned)(uintptr_t)&feat[m][256 + l * 4];
            asm volatile("global_load_async_to_lds_b128 %0, %1, off"
                         :: "v"(l0), "v"(gx) : "memory");
            asm volatile("global_load_async_to_lds_b128 %0, %1, off"
                         :: "v"(l1), "v"(gd) : "memory");
            asm volatile("global_load_async_to_lds_b128 %0, %1, off"
                         :: "v"(l2), "v"(ge) : "memory");
        } else {
            for (int c = l; c < 384; c += 32) feat[m][c] = 0.0f;
        }
    }
    asm volatile("s_wait_asynccnt 0x0" ::: "memory");   // async DMA -> LDS visible
    __syncthreads();

    const int mrow = l & 15;
    const int hi   = l >> 4;                 // 0: lanes 0-15, 1: lanes 16-31
    union { v16h v; _Float16 e[16]; } A, B;
    union { v8f  v; float    f[8];  } acc;
    acc.v = (v8f){};

    #pragma unroll
    for (int kb = 0; kb < 384; kb += 32) {
        #pragma unroll
        for (int j = 0; j < 16; ++j) {
            int ka = (j < 8 ? j : j + 8) + hi * 8;       // documented A layout
            A.e[j] = (_Float16)feat[mrow][kb + ka];
            B.e[j] = (_Float16)lw[kb + ka];     // B[k][n]=w[k] for all n
        }
        acc.v = __builtin_amdgcn_wmma_f32_16x16x32_f16(
            false, A.v, false, B.v, (short)0, acc.v, false, false);
    }
    // D layout: lane l, vgpr r -> M = r + 8*hi, N = l%16. Column 0 carries dots.
    if (mrow == 0) {
        float bias = lb[0];
        #pragma unroll
        for (int r = 0; r < 8; ++r) {
            int e = base + r + hi * 8;
            if (e < E) raw[e] = acc.f[r] + bias;
        }
    }
}

// scalar fallback (only used when C != 128)
__global__ void score_scalar(const float* x, const int* ei, const float* ea,
                             const float* lw, const float* lb, float* raw,
                             int C, int E) {
    int e = blockIdx.x * blockDim.x + threadIdx.x;
    if (e >= E) return;
    int s = ei[e], d = ei[E + e];
    float acc = lb[0];
    for (int c = 0; c < C; ++c) acc += x[s * C + c] * lw[c];
    for (int c = 0; c < C; ++c) acc += x[d * C + c] * lw[C + c];
    for (int c = 0; c < C; ++c) acc += ea[e * C + c] * lw[2 * C + c];
    raw[e] = acc;
}

// ---------- 2-4. segment softmax over dst ----------
__global__ void seg_max(const float* raw, const int* ei, unsigned* mmax, int E) {
    int e = blockIdx.x * blockDim.x + threadIdx.x;
    if (e >= E) return;
    atomicMax(&mmax[ei[E + e]], f2o(raw[e]));
}
__global__ void seg_expsum(const float* raw, const int* ei, const unsigned* mmax,
                           float* ssum, float* score, int E) {
    int e = blockIdx.x * blockDim.x + threadIdx.x;
    if (e >= E) return;
    int d = ei[E + e];
    float ex = __expf(raw[e] - o2f(mmax[d]));
    score[e] = ex;
    atomicAdd(&ssum[d], ex);
}
__global__ void seg_norm(float* score, const int* ei, const float* ssum, int E) {
    int e = blockIdx.x * blockDim.x + threadIdx.x;
    if (e >= E) return;
    score[e] = score[e] / ssum[ei[E + e]] + 0.5f;     // ADD_TO_EDGE_SCORE
}

// ---------- 5. stable sort by descending score: key=(~ord(score)<<32)|idx ----------
__global__ void build_key_score(unsigned long long* key, const float* score,
                                int E, int E2) {
    int i = blockIdx.x * blockDim.x + threadIdx.x;
    if (i >= E2) return;
    key[i] = (i < E) ? (((unsigned long long)(~f2o(score[i])) << 32) | (unsigned)i)
                     : ~0ull;
}
__global__ void bitonic_pass(unsigned long long* key, int j, int k) {
    int i = blockIdx.x * blockDim.x + threadIdx.x;
    int ixj = i ^ j;
    if (ixj > i) {
        unsigned long long a = key[i], b = key[ixj];
        bool up = ((i & k) == 0);
        if ((a > b) == up) { key[i] = b; key[ixj] = a; }
    }
}

// ---------- 6. sequential greedy maximal matching ----------
__global__ void greedy_match(const unsigned long long* key, const int* ei,
                             const float* score, int* rem, int* clus,
                             float* nes, int* scal, int E) {
    if (threadIdx.x != 0 || blockIdx.x != 0) return;
    int nm = 0;
    for (int p = 0; p < E; ++p) {
        __builtin_prefetch(&key[p + 16], 0, 0);
        unsigned idx = (unsigned)(key[p] & 0xFFFFFFFFu);
        int s = ei[idx], d = ei[E + idx];
        if (rem[s] && rem[d]) {
            clus[s] = nm; clus[d] = nm;
            rem[s] = 0;   rem[d] = 0;
            nes[nm] = score[idx];
            ++nm;
        }
    }
    scal[0] = nm;
}
__global__ void assign_singletons(const int* rem, int* clus, int* scal,
                                  float* o_scal, int N) {
    if (threadIdx.x != 0 || blockIdx.x != 0) return;
    int nm = scal[0], r = 0;
    for (int v = 0; v < N; ++v)
        if (rem[v]) clus[v] = nm + (r++);
    scal[1] = nm + r;
    o_scal[0] = (float)(nm + r);                      // num_clusters
}

// ---------- 7. pooled node features ----------
__global__ void scatter_newx(const float* x, const int* clus, float* o_x,
                             int N, int C) {
    int i = blockIdx.x * blockDim.x + threadIdx.x;
    if (i >= N * C) return;
    int v = i / C, c = i - v * C;
    atomicAdd(&o_x[clus[v] * C + c], x[i]);
}
__global__ void scale_newx(float* o_x, const float* nes, int N, int C) {
    int i = blockIdx.x * blockDim.x + threadIdx.x;
    if (i >= N * C) return;
    o_x[i] *= nes[i / C];
}
__global__ void batch_cluster_out(const int* batch, const int* clus,
                                  float* o_batch, float* o_clus, int N) {
    int v = blockIdx.x * blockDim.x + threadIdx.x;
    if (v >= N) return;
    o_batch[clus[v]] = (float)batch[v];
    o_clus[v] = (float)clus[v];
}

// ---------- 8. coalesce pooled edges ----------
__global__ void build_key_coalesce(unsigned long long* key, const int* ei,
                                   const int* clus, int N, int E, int E2) {
    int i = blockIdx.x * blockDim.x + threadIdx.x;
    if (i >= E2) return;
    if (i < E) {
        unsigned id = (unsigned)clus[ei[i]] * (unsigned)N + (unsigned)clus[ei[E + i]];
        key[i] = ((unsigned long long)id << 32) | (unsigned)i;
    } else key[i] = ~0ull;
}
__global__ void coalesce_walk(const unsigned long long* key, unsigned* segid,
                              float* o_row, float* o_col, int* scal,
                              float* o_scal, int N, int E) {
    if (threadIdx.x != 0 || blockIdx.x != 0) return;
    int seg = -1;
    unsigned prev = 0xFFFFFFFFu;
    for (int p = 0; p < E; ++p) {
        __builtin_prefetch(&key[p + 16], 0, 0);
        unsigned long long kk = key[p];
        unsigned id = (unsigned)(kk >> 32);
        unsigned oi = (unsigned)(kk & 0xFFFFFFFFu);
        if (seg < 0 || id != prev) {
            ++seg; prev = id;
            o_row[seg] = (float)(id / (unsigned)N);
            o_col[seg] = (float)(id % (unsigned)N);
        }
        segid[oi] = (unsigned)seg;
    }
    scal[2] = seg + 1;
    o_scal[1] = (float)(seg + 1);                     // num_new_edges
}
__global__ void scatter_newattr(const float* ea, const unsigned* segid,
                                float* o_ea, int E, int C) {
    long long i = (long long)blockIdx.x * blockDim.x + threadIdx.x;
    if (i >= (long long)E * C) return;
    int e = (int)(i / C), c = (int)(i - (long long)e * C);
    atomicAdd(&o_ea[(long long)segid[e] * C + c], ea[i]);
}

// ---------- host launcher ----------
extern "C" void kernel_launch(void* const* d_in, const int* in_sizes, int n_in,
                              void* d_out, int out_size, void* d_ws, size_t ws_size,
                              hipStream_t stream) {
    const float* x     = (const float*)d_in[0];
    const int*   ei    = (const int*)  d_in[1];
    const float* ea    = (const float*)d_in[2];
    const int*   batch = (const int*)  d_in[3];
    const float* lw    = (const float*)d_in[4];
    const float* lb    = (const float*)d_in[5];

    const int N = in_sizes[3];
    const int C = in_sizes[4] / 3;
    const int E = in_sizes[1] / 2;
    int E2 = 1; while (E2 < E) E2 <<= 1;

    // workspace carve-up (256B aligned regions)
    char* w = (char*)d_ws;
    auto carve = [&](size_t bytes) { char* p = w; w += (bytes + 255) & ~size_t(255); return p; };
    float*              raw   = (float*)             carve(sizeof(float) * E);
    unsigned*           mmax  = (unsigned*)          carve(sizeof(unsigned) * N);
    float*              ssum  = (float*)             carve(sizeof(float) * N);
    float*              score = (float*)             carve(sizeof(float) * E);
    unsigned long long* key   = (unsigned long long*)carve(sizeof(unsigned long long) * E2);
    unsigned*           segid = (unsigned*)          carve(sizeof(unsigned) * E);
    int*                clus  = (int*)               carve(sizeof(int) * N);
    int*                rem   = (int*)               carve(sizeof(int) * N);
    int*                scal  = (int*)               carve(sizeof(int) * 16);

    // output carve-up (reference tuple order, all as f32)
    float* o_x     = (float*)d_out;          // [N, C]
    float* o_row   = o_x + (size_t)N * C;    // new_edge_index row 0  [E]
    float* o_col   = o_row + E;              // new_edge_index row 1  [E]
    float* o_ea    = o_col + E;              // [E, C]
    float* o_batch = o_ea + (size_t)E * C;   // [N]
    float* o_nes   = o_batch + N;            // [N]
    float* o_clus  = o_nes + N;              // [N]
    float* o_scal  = o_clus + N;             // [2]

    const int B = 256;
    // 0) clear output + init per-node state (nes -> 1.0 after clear)
    zero_f32<<<2048, B, 0, stream>>>((float*)d_out, (long long)out_size);
    init_node_state<<<(N + B - 1) / B, B, 0, stream>>>(mmax, ssum, rem, clus, o_nes, N);

    // 1) edge scores via WMMA + async-to-LDS staging (C==128 fast path)
    if (C == 128)
        score_wmma<<<(E + 15) / 16, 32, 0, stream>>>(x, ei, ea, lw, lb, raw, E);
    else
        score_scalar<<<(E + B - 1) / B, B, 0, stream>>>(x, ei, ea, lw, lb, raw, C, E);

    // 2-4) segment softmax over dst, +0.5
    seg_max   <<<(E + B - 1) / B, B, 0, stream>>>(raw, ei, mmax, E);
    seg_expsum<<<(E + B - 1) / B, B, 0, stream>>>(raw, ei, mmax, ssum, score, E);
    seg_norm  <<<(E + B - 1) / B, B, 0, stream>>>(score, ei, ssum, E);

    // 5) stable bitonic sort by descending score
    build_key_score<<<(E2 + B - 1) / B, B, 0, stream>>>(key, score, E, E2);
    for (int k = 2; k <= E2; k <<= 1)
        for (int j = k >> 1; j > 0; j >>= 1)
            bitonic_pass<<<(E2 + B - 1) / B, B, 0, stream>>>(key, j, k);

    // 6) sequential greedy matching + singleton clusters
    greedy_match<<<1, 32, 0, stream>>>(key, ei, score, rem, clus, o_nes, scal, E);
    assign_singletons<<<1, 32, 0, stream>>>(rem, clus, scal, o_scal, N);

    // 7) pooled node features / batch / cluster outputs
    scatter_newx<<<((size_t)N * C + B - 1) / B, B, 0, stream>>>(x, clus, o_x, N, C);
    scale_newx  <<<((size_t)N * C + B - 1) / B, B, 0, stream>>>(o_x, o_nes, N, C);
    batch_cluster_out<<<(N + B - 1) / B, B, 0, stream>>>(batch, clus, o_batch, o_clus, N);

    // 8) coalesce pooled connectivity
    build_key_coalesce<<<(E2 + B - 1) / B, B, 0, stream>>>(key, ei, clus, N, E, E2);
    for (int k = 2; k <= E2; k <<= 1)
        for (int j = k >> 1; j > 0; j >>= 1)
            bitonic_pass<<<(E2 + B - 1) / B, B, 0, stream>>>(key, j, k);
    coalesce_walk<<<1, 32, 0, stream>>>(key, segid, o_row, o_col, scal, o_scal, N, E);
    scatter_newattr<<<(int)(((long long)E * C + B - 1) / B), B, 0, stream>>>(ea, segid, o_ea, E, C);
}
